// BilinearAttention_70600672412000
// MI455X (gfx1250) — matrix operand order repaired and checked by reading, not compiled, not run on hardware
//
#include <hip/hip_runtime.h>
#include <hip/hip_bf16.h>

// ---------------------------------------------------------------------------
// BilinearAttention (BAN-style) for MI455X / gfx1250, bf16 WMMA path with
// GLOBAL_LOAD_ASYNC_TO_LDS_B128 staging.
// B=32, N=128, L=1024, DRUG_DIM=256, TGT_DIM=1280, HID=512, H=8, HD=64
// ---------------------------------------------------------------------------

typedef __bf16 bhalf;
typedef __attribute__((ext_vector_type(16))) __bf16 v16bf;
typedef __attribute__((ext_vector_type(8)))  __bf16 v8bf;
typedef __attribute__((ext_vector_type(4)))  __bf16 v4bf;
typedef __attribute__((ext_vector_type(8)))  float  v8f;
typedef __attribute__((ext_vector_type(4)))  float  v4f;

#define NEGV (-1.0e9f)

// ---- CDNA5 async global->LDS copy (16 bytes / lane), ASYNCcnt-tracked -----
__device__ __forceinline__ void async_copy16(const void* gptr, void* lptr) {
  unsigned int  loff = (unsigned int)(unsigned long long)lptr;  // LDS = addr[31:0]
  unsigned long long ga = (unsigned long long)gptr;
  asm volatile("global_load_async_to_lds_b128 %0, %1, off"
               :: "v"(loff), "v"(ga) : "memory");
}
__device__ __forceinline__ void async_wait0() {
  asm volatile("s_wait_asynccnt 0x0" ::: "memory");
}

// ---- WMMA fragment loaders (LDS, row-major, ld in elements) ---------------
// A matrix 16x32 bf16: lanes 0-15 -> row m=lane, K {0..7,16..23};
//                      lanes 16-31 -> row m=lane-16, K {8..15,24..31}.
__device__ __forceinline__ v16bf load_fragA(const bhalf* base, int ld, int row0, int k0) {
  const int lane = threadIdx.x & 31;
  const int r  = row0 + (lane & 15);
  const int ko = k0 + ((lane >> 4) << 3);
  const bhalf* p = base + r * ld + ko;
  v8bf lo = *(const v8bf*)(p);
  v8bf hi = *(const v8bf*)(p + 16);
  return __builtin_shufflevector(lo, hi, 0,1,2,3,4,5,6,7,8,9,10,11,12,13,14,15);
}

// B matrix 32x16 bf16 (fed as B^T rows = columns of B):
// lanes 0-15 -> col n=lane, K {0..7,16..23}; lanes 16-31 -> col n=lane-16, K {8..15,24..31}.
__device__ __forceinline__ v16bf load_fragB(const bhalf* base, int ld, int col0, int k0) {
  const int lane = threadIdx.x & 31;
  const int r  = col0 + (lane & 15);
  const int ko = k0 + ((lane >> 4) << 3);
  const bhalf* p = base + r * ld + ko;
  v8bf lo = *(const v8bf*)(p);
  v8bf hi = *(const v8bf*)(p + 16);
  return __builtin_shufflevector(lo, hi, 0,1,2,3,4,5,6,7,8,9,10,11,12,13,14,15);
}

__device__ __forceinline__ v8f wmma_bf16(v16bf a, v16bf b, v8f c) {
  // 8 args: (neg_a, A, neg_b, B, c_mod, C, reuse_a, reuse_b)
  return __builtin_amdgcn_wmma_f32_16x16x32_bf16(false, a, false, b, (short)0, c, false, false);
}

// ---------------------------------------------------------------------------
// Weight prep: fp32 -> bf16 with transpose (so B-operand rows are columns)
// ---------------------------------------------------------------------------
__global__ void transpose_to_bf16(const float* __restrict__ src, bhalf* __restrict__ dst,
                                  int R, int C) {
  int total = R * C;
  for (int i = blockIdx.x * blockDim.x + threadIdx.x; i < total; i += gridDim.x * blockDim.x) {
    int r = i / C, c = i % C;
    dst[(size_t)c * R + r] = (bhalf)src[i];
  }
}

__global__ void transpose_wb_bf16(const float* __restrict__ src, bhalf* __restrict__ dst) {
  for (int i = blockIdx.x * blockDim.x + threadIdx.x; i < 8 * 64 * 64; i += gridDim.x * blockDim.x) {
    int h = i >> 12, rem = i & 4095, e = rem >> 6, d = rem & 63;
    dst[(h << 12) + (e << 6) + d] = (bhalf)src[(h << 12) + (d << 6) + e];
  }
}

// ---------------------------------------------------------------------------
// Projection GEMM: O[M,512](bf16) = X[M,K](f32) @ W + bias, W given as
// WT (512 rows x K) bf16. Block = 256 thr (8 waves); tile 128(M) x 64(N).
// A tile: vectorized f32->bf16 convert; B tile: async b128 LDS copy.
// ---------------------------------------------------------------------------
__global__ __launch_bounds__(256) void proj_kernel(const float* __restrict__ X,
                                                   const bhalf* __restrict__ WT,
                                                   const float* __restrict__ bias,
                                                   bhalf* __restrict__ O,
                                                   int M, int K) {
  __shared__ alignas(16) bhalf At[128 * 32];
  __shared__ alignas(16) bhalf Bt[64 * 32];
  const int tid = threadIdx.x, wave = tid >> 5, lane = tid & 31;
  const int col0 = blockIdx.x * 64, row0 = blockIdx.y * 128;

  v8f acc[4] = {};
  for (int k0 = 0; k0 < K; k0 += 32) {
    // B tile: 64 rows x 32 bf16 = 4 KB, one 16B async chunk per thread
    {
      int r = tid >> 2, c = tid & 3;
      async_copy16(WT + (size_t)(col0 + r) * K + k0 + c * 8, Bt + r * 32 + c * 8);
    }
    // A tile: 128 rows x 32 f32 -> bf16, 4 x float4 per thread
#pragma unroll
    for (int j = 0; j < 4; ++j) {
      int i = tid + j * 256;          // 1024 chunks of 4 floats
      int r = i >> 3, c4 = i & 7;
      v4f x = *(const v4f*)(X + (size_t)(row0 + r) * K + k0 + c4 * 4);
      v4bf y;
      y.x = (bhalf)x.x; y.y = (bhalf)x.y; y.z = (bhalf)x.z; y.w = (bhalf)x.w;
      *(v4bf*)(At + r * 32 + c4 * 4) = y;
    }
    async_wait0();
    __syncthreads();

    v16bf a = load_fragA(At, 32, wave * 16, 0);
#pragma unroll
    for (int nt = 0; nt < 4; ++nt) {
      v16bf b = load_fragB(Bt, 32, nt * 16, 0);
      acc[nt] = wmma_bf16(a, b, acc[nt]);
    }
    __syncthreads();
  }
#pragma unroll
  for (int nt = 0; nt < 4; ++nt) {
#pragma unroll
    for (int r = 0; r < 8; ++r) {
      int row = row0 + wave * 16 + r + ((lane >> 4) << 3);
      int col = col0 + nt * 16 + (lane & 15);
      O[(size_t)row * 512 + col] = (bhalf)(acc[nt][r] + bias[col]);
    }
  }
}

// ---------------------------------------------------------------------------
// Fused bilinear attention per (b,h). Streams L in 64-wide tiles, two passes.
// Pass 1: running row softmax stats (m,s over L) + per-tile column softmax
//         accumulating w[n] = sum_{l valid} attn_t[n,l].
// Pass 2: recompute scores, v[l] = sum_{n valid} attn_d[n,l], accumulate
//         acc_d[e] += sum_l th[l,e]*v[l].
// Final:  out_d = acc_d/Nv ; out_t[e] = (sum_n dh[n,e]*w[n]) / Lv.
// ---------------------------------------------------------------------------
struct SMem {
  alignas(16) bhalf aw[128 * 64];   // dWb = dh @ Wb[h], bf16
  alignas(16) bhalf th[64 * 64];    // current target tile, bf16
  union {
    alignas(16) float S[128 * 68];  // score tile (padded stride 68)
    struct {
      alignas(16) bhalf dh[128 * 64];
      alignas(16) bhalf wb[64 * 64];
    } p1;
  } u;
  float rowm[128], rowsum[128], roww[128];
  float colmax[64], colsum[64], vcol[64];
  float accd[64];
  unsigned char dm[128];
  unsigned char tmt[64];
  int nv, lv;
};

__device__ __forceinline__ void compute_s_tile(SMem& sm, int wave, int lane) {
  const int rb = wave * 16;
  v16bf a0 = load_fragA(sm.aw, 64, rb, 0);
  v16bf a1 = load_fragA(sm.aw, 64, rb, 32);
#pragma unroll
  for (int ct = 0; ct < 4; ++ct) {
    v16bf b0 = load_fragB(sm.th, 64, ct * 16, 0);
    v16bf b1 = load_fragB(sm.th, 64, ct * 16, 32);
    v8f cc = {};
    cc = wmma_bf16(a0, b0, cc);
    cc = wmma_bf16(a1, b1, cc);
#pragma unroll
    for (int r = 0; r < 8; ++r) {
      int row = rb + r + ((lane >> 4) << 3);
      int col = ct * 16 + (lane & 15);
      bool mv = sm.dm[row] && sm.tmt[col];
      sm.u.S[row * 68 + col] = mv ? cc[r] : NEGV;
    }
  }
}

__global__ __launch_bounds__(256) void attn_kernel(const bhalf* __restrict__ Dproj,
                                                   const bhalf* __restrict__ Tproj,
                                                   const bhalf* __restrict__ WbT,
                                                   const unsigned char* __restrict__ dmask,
                                                   const unsigned char* __restrict__ tmask,
                                                   float* __restrict__ out) {
  __shared__ SMem sm;
  const int tid = threadIdx.x, wave = tid >> 5, lane = tid & 31;
  const int b = blockIdx.x >> 3, h = blockIdx.x & 7;
  const bhalf* dgl = Dproj + (size_t)b * 128 * 512 + h * 64;   // row stride 512
  const bhalf* tgl = Tproj + (size_t)b * 1024 * 512 + h * 64;  // row stride 512
  const bhalf* wbp = WbT + h * 4096;

  // stage dh (16 KB) and Wb^T[h] (8 KB) via async b128 copies
#pragma unroll
  for (int j = 0; j < 4; ++j) {
    int i = tid + j * 256;            // 1024 chunks: 128 rows x 8
    int n = i >> 3, c = i & 7;
    async_copy16(dgl + (size_t)n * 512 + c * 8, sm.u.p1.dh + n * 64 + c * 8);
  }
#pragma unroll
  for (int j = 0; j < 2; ++j) {
    int i = tid + j * 256;            // 512 contiguous chunks
    async_copy16(wbp + i * 8, sm.u.p1.wb + i * 8);
  }
  if (tid < 128) sm.dm[tid] = dmask[b * 128 + tid];
  async_wait0();
  __syncthreads();

  if (tid == 0) {
    int s = 0;
    for (int n = 0; n < 128; ++n) s += sm.dm[n];
    sm.nv = (s > 0) ? s : 1;
  }
  if (tid == 32) {
    int s = 0;
    const unsigned char* tp = tmask + b * 1024;
    for (int l = 0; l < 1024; ++l) s += tp[l];
    sm.lv = (s > 0) ? s : 1;
  }

  // dWb = dh @ Wb[h]  (128x64, K=64), result bf16 into sm.aw
  {
    const int rb = wave * 16;
    v16bf a0 = load_fragA(sm.u.p1.dh, 64, rb, 0);
    v16bf a1 = load_fragA(sm.u.p1.dh, 64, rb, 32);
#pragma unroll
    for (int nt = 0; nt < 4; ++nt) {
      v16bf b0 = load_fragB(sm.u.p1.wb, 64, nt * 16, 0);
      v16bf b1 = load_fragB(sm.u.p1.wb, 64, nt * 16, 32);
      v8f cc = {};
      cc = wmma_bf16(a0, b0, cc);
      cc = wmma_bf16(a1, b1, cc);
#pragma unroll
      for (int r = 0; r < 8; ++r) {
        int row = rb + r + ((lane >> 4) << 3);
        int col = nt * 16 + (lane & 15);
        sm.aw[row * 64 + col] = (bhalf)cc[r];
      }
    }
  }
  if (tid < 128) { sm.rowm[tid] = -1e30f; sm.rowsum[tid] = 0.f; sm.roww[tid] = 0.f; }
  if (tid < 64) sm.accd[tid] = 0.f;
  __syncthreads();

  // ---------------- pass 1 ----------------
  for (int lt = 0; lt < 16; ++lt) {
    const int l0 = lt * 64;
#pragma unroll
    for (int j = 0; j < 2; ++j) {     // th tile: 512 chunks of 16B
      int i = tid + j * 256;
      int l = i >> 3, c = i & 7;
      async_copy16(tgl + (size_t)(l0 + l) * 512 + c * 8, sm.th + l * 64 + c * 8);
    }
    if (tid < 64) sm.tmt[tid] = tmask[b * 1024 + l0 + tid];
    async_wait0();
    __syncthreads();
    compute_s_tile(sm, wave, lane);
    __syncthreads();
    if (tid < 64) {  // column softmax stats over n (full N=128 in-tile)
      const int c = tid;
      float cm = -1e30f;
      for (int n = 0; n < 128; ++n) cm = fmaxf(cm, sm.u.S[n * 68 + c]);
      float cs = 0.f;
      for (int n = 0; n < 128; ++n) cs += __expf(sm.u.S[n * 68 + c] - cm);
      sm.colmax[c] = cm; sm.colsum[c] = cs;
    }
    __syncthreads();
    if (tid < 128 && sm.dm[tid]) {  // row running stats + w accumulation
      const int n = tid;
      const float* Sr = &sm.u.S[n * 68];
      float tmx = -1e30f;
      for (int c = 0; c < 64; ++c) tmx = fmaxf(tmx, Sr[c]);
      float mn = fmaxf(sm.rowm[n], tmx);
      float ss = 0.f, wacc = 0.f;
      for (int c = 0; c < 64; ++c) {
        float sv = Sr[c];
        ss += __expf(sv - mn);
        if (sm.tmt[c]) wacc += __expf(sv - sm.colmax[c]) / sm.colsum[c];
      }
      sm.rowsum[n] = sm.rowsum[n] * __expf(sm.rowm[n] - mn) + ss;
      sm.rowm[n] = mn;
      sm.roww[n] += wacc;
    }
    __syncthreads();
  }

  // ---------------- pass 2 ----------------
  for (int lt = 0; lt < 16; ++lt) {
    const int l0 = lt * 64;
#pragma unroll
    for (int j = 0; j < 2; ++j) {
      int i = tid + j * 256;
      int l = i >> 3, c = i & 7;
      async_copy16(tgl + (size_t)(l0 + l) * 512 + c * 8, sm.th + l * 64 + c * 8);
    }
    if (tid < 64) sm.tmt[tid] = tmask[b * 1024 + l0 + tid];
    async_wait0();
    __syncthreads();
    compute_s_tile(sm, wave, lane);
    __syncthreads();
    if (tid < 64) {  // v[l] = sum over valid n of attn_d
      const int c = tid;
      float v = 0.f;
      if (sm.tmt[c]) {
        for (int n = 0; n < 128; ++n)
          if (sm.dm[n]) v += __expf(sm.u.S[n * 68 + c] - sm.rowm[n]) / sm.rowsum[n];
      }
      sm.vcol[c] = v;
    }
    __syncthreads();
    if (tid < 64) {  // acc_d[e] += th^T @ v
      const int e = tid;
      float a = 0.f;
      for (int c = 0; c < 64; ++c) a += (float)sm.th[c * 64 + e] * sm.vcol[c];
      sm.accd[e] += a;
    }
    __syncthreads();
  }

  // ---------------- final pooled outputs ----------------
  if (tid < 64) {
    const int e = tid;
    float at = 0.f;
    for (int n = 0; n < 128; ++n) at += (float)dgl[(size_t)n * 512 + e] * sm.roww[n];
    out[(size_t)b * 1024 + h * 64 + e]       = sm.accd[e] / (float)sm.nv;
    out[(size_t)b * 1024 + 512 + h * 64 + e] = at / (float)sm.lv;
  }
}

// ---------------------------------------------------------------------------
// Host launcher
// ---------------------------------------------------------------------------
extern "C" void kernel_launch(void* const* d_in, const int* in_sizes, int n_in,
                              void* d_out, int out_size, void* d_ws, size_t ws_size,
                              hipStream_t stream) {
  (void)in_sizes; (void)n_in; (void)out_size; (void)ws_size;
  const float*         drug_nodes  = (const float*)d_in[0];
  const unsigned char* drug_mask   = (const unsigned char*)d_in[1];
  const float*         target_seq  = (const float*)d_in[2];
  const unsigned char* target_mask = (const unsigned char*)d_in[3];
  const float*         Wd          = (const float*)d_in[4];
  const float*         bd          = (const float*)d_in[5];
  const float*         Wt          = (const float*)d_in[6];
  const float*         bt          = (const float*)d_in[7];
  const float*         Wb          = (const float*)d_in[8];
  float* out = (float*)d_out;

  char* ws = (char*)d_ws;
  size_t off = 0;
  bhalf* WdT = (bhalf*)(ws + off); off += (size_t)512 * 256 * sizeof(bhalf);
  bhalf* WtT = (bhalf*)(ws + off); off += (size_t)512 * 1280 * sizeof(bhalf);
  bhalf* WbT = (bhalf*)(ws + off); off += (size_t)8 * 64 * 64 * sizeof(bhalf);
  bhalf* Dp  = (bhalf*)(ws + off); off += (size_t)4096 * 512 * sizeof(bhalf);
  bhalf* Tp  = (bhalf*)(ws + off); off += (size_t)32768 * 512 * sizeof(bhalf);

  // weight prep (fp32 -> bf16, transposed)
  transpose_to_bf16<<<dim3(256), dim3(256), 0, stream>>>(Wd, WdT, 256, 512);
  transpose_to_bf16<<<dim3(512), dim3(256), 0, stream>>>(Wt, WtT, 1280, 512);
  transpose_wb_bf16<<<dim3(64), dim3(256), 0, stream>>>(Wb, WbT);

  // projections (bf16 WMMA, fp32 accumulate)
  proj_kernel<<<dim3(8, 32),  dim3(256), 0, stream>>>(drug_nodes, WdT, bd, Dp, 4096, 256);
  proj_kernel<<<dim3(8, 256), dim3(256), 0, stream>>>(target_seq, WtT, bt, Tp, 32768, 1280);

  // fused bilinear attention + dual softmax + masked mean pooling
  attn_kernel<<<dim3(32 * 8), dim3(256), 0, stream>>>(Dp, Tp, WbT, drug_mask, target_mask, out);
}
